// CNN_61014305407527
// MI455X (gfx1250) — matrix operand order repaired and checked by reading, not compiled, not run on hardware
//
#include <hip/hip_runtime.h>
#include <hip/hip_bf16.h>

// ---------------- problem constants (from reference) ----------------
#define Bn    512
#define Sn    128
#define En    300
#define NFn   100
#define NCLS  5

// padded geometry for WMMA tiling
#define EP    320          // E padded to multiple of 32 (K dim, global layout)
#define EPL   328          // LDS row stride (bank-conflict padding, 656 B)
#define S4    (Sn + 4)     // 2 zero halo rows each side of the time axis
#define ECH   (EP / 32)    // 10 K-chunks of 32 per window row
#define NFT   7            // ceil(100/16) filter tiles

// LDS staging: 68 rows cover one half (64 t positions + max halo of 4)
#define STG_ROWS   68
#define STG_CHUNKS (STG_ROWS * (EP / 8))   // 16-byte chunks = 2720

#define USE_ASYNC_LDS 1

typedef __attribute__((ext_vector_type(16))) __bf16 v16bf;
typedef __attribute__((ext_vector_type(8)))  __bf16 v8bf;
typedef __attribute__((ext_vector_type(8)))  float  v8f;

// ---------------------------------------------------------------
// 1) fp32 x [B,S,E] -> bf16 xpad [B,S4,EP], zero halo rows + E pad
// ---------------------------------------------------------------
__global__ void cvt_x_kernel(const float* __restrict__ x,
                             __bf16* __restrict__ xpad, int total) {
  int i = blockIdx.x * blockDim.x + threadIdx.x;
  if (i >= total) return;
  int e = i % EP;
  int r = (i / EP) % S4;
  int b = i / (EP * S4);
  int t = r - 2;
  float v = 0.0f;
  if (t >= 0 && t < Sn && e < En)
    v = x[((size_t)b * Sn + t) * En + e];
  xpad[i] = (__bf16)v;
}

// ----------------------------------------------------------------------
// 2) pack conv weights into per-lane WMMA B-fragment layout (bf16).
//    block id = (row*ECH + ec)*NFT + fTile, row = cumulative (branch,dk).
//    lane L holds N = L&15, K = (L<16 ? 0..7,16..23 : 8..15,24..31): 16 bf16.
// ----------------------------------------------------------------------
__global__ void pack_w_kernel(const float* __restrict__ w3,
                              const float* __restrict__ w4,
                              const float* __restrict__ w5,
                              __bf16* __restrict__ wfrag) {
  int blk  = blockIdx.x;            // 0 .. 12*ECH*NFT-1
  int lane = threadIdx.x;           // 0..31
  int fTile = blk % NFT;
  int ec    = (blk / NFT) % ECH;
  int row   = blk / (NFT * ECH);    // 0..11
  int branch, dk;
  if (row < 3)      { branch = 0; dk = row;     }
  else if (row < 7) { branch = 1; dk = row - 3; }
  else              { branch = 2; dk = row - 7; }
  const float* w = (branch == 0) ? w3 : (branch == 1) ? w4 : w5;
  int k  = 3 + branch;
  int f  = fTile * 16 + (lane & 15);
  int e0 = ec * 32 + ((lane < 16) ? 0 : 8);
  __bf16* dst = wfrag + (size_t)blk * 512 + (size_t)lane * 16;
#pragma unroll
  for (int j = 0; j < 16; ++j) {
    int e = e0 + j + ((j < 8) ? 0 : 8);
    float v = 0.0f;
    if (f < NFn && e < En)
      v = w[(size_t)f * (k * En) + (size_t)dk * En + e];
    dst[j] = (__bf16)v;
  }
}

// ---------------------------------------------------------------
// 3) zero the pooled-features accumulator (atomic-max target)
// ---------------------------------------------------------------
__global__ void zero_feats_kernel(float* __restrict__ feats, int n) {
  int i = blockIdx.x * blockDim.x + threadIdx.x;
  if (i < n) feats[i] = 0.0f;
}

// ----------------------------------------------------------------------
// 4) conv-as-GEMM: async-stage x tile to LDS, WMMA bf16 from ds_load,
//    fused bias+ReLU+global-max epilogue.  One instantiation per branch.
//    grid = (B, 2 halves), block = 256 (8 waves). Wave round-robins over
//    the 4 t-tiles x 7 f-tiles of its half.
// ----------------------------------------------------------------------
template <int K, int PAD, int ROWB>
__global__ void __launch_bounds__(256)
conv_wmma_kernel(const __bf16* __restrict__ xpad,
                 const __bf16* __restrict__ wfrag,
                 const float* __restrict__ bias,
                 float* __restrict__ feats, int foff) {
  constexpr int Sp = Sn + 2 * PAD - K + 1;     // valid output length
  __shared__ __align__(16) __bf16 tile[STG_ROWS * EPL];  // 44,608 B

  const int b    = blockIdx.x;
  const int half = blockIdx.y;                 // 0: t 0..63, 1: t 64..127
  const int rowBase0 = half * 64;              // first staged global row
  const int lane = threadIdx.x & 31;
  const int wave = threadIdx.x >> 5;           // 0..7

  // ---- stage 68 x EP rows of this batch element into LDS (async DMA) ----
  {
    const __bf16* srcBase = xpad + ((size_t)b * S4 + rowBase0) * EP;
    for (int c = threadIdx.x; c < STG_CHUNKS; c += 256) {
      int r  = c / (EP / 8);                   // row 0..67
      int cc = c - r * (EP / 8);               // 16B chunk within row
      const __bf16* gp = srcBase + (size_t)r * EP + cc * 8;
      __bf16* lpp = tile + r * EPL + cc * 8;
#if USE_ASYNC_LDS
      unsigned lp = (unsigned)(size_t)lpp;     // low 32 bits = LDS byte addr
      asm volatile("global_load_async_to_lds_b128 %0, %1, off"
                   :: "v"(lp), "v"(gp) : "memory");
#else
      *(v8bf*)lpp = *(const v8bf*)gp;
#endif
    }
#if USE_ASYNC_LDS
    asm volatile("s_wait_asynccnt 0" ::: "memory");
#endif
    __syncthreads();
  }

  const int m     = lane & 15;                 // A row / C column index
  const int kOffA = (lane < 16) ? 0 : 8;       // A-fragment K-half offset

  // ---- 4 t-tiles x 7 f-tiles, round-robined across the 8 waves ----
  for (int idx = wave; idx < 4 * NFT; idx += 8) {
    const int tTile = idx & 3;
    const int fTile = idx >> 2;
    const int tBase = rowBase0 + tTile * 16;
    // lane's LDS row at dk=0 (halo zeros already staged)
    const __bf16* arow0 = tile + (tTile * 16 + m - PAD + 2) * EPL + kOffA;
    const __bf16* wrow0 = wfrag + ((size_t)ROWB * ECH * NFT + fTile) * 512
                                + (size_t)lane * 16;
    v8f acc = {};
#pragma unroll
    for (int dk = 0; dk < K; ++dk) {
      const __bf16* arow = arow0 + dk * EPL;
      const __bf16* wrow = wrow0 + (size_t)dk * ECH * NFT * 512;
      __builtin_prefetch(wrow + (size_t)ECH * NFT * 512, 0, 0);
#pragma unroll
      for (int ec = 0; ec < ECH; ++ec) {
        v8bf lo = *(const v8bf*)(arow + ec * 32);        // K 0..7  / 8..15
        v8bf hi = *(const v8bf*)(arow + ec * 32 + 16);   // K 16..23/ 24..31
        v16bf a = __builtin_shufflevector(lo, hi,
                   0, 1, 2, 3, 4, 5, 6, 7, 8, 9, 10, 11, 12, 13, 14, 15);
        v16bf bw = *(const v16bf*)(wrow + (size_t)ec * NFT * 512);
        acc = __builtin_amdgcn_wmma_f32_16x16x32_bf16(
                  false, a, false, bw, (short)0, acc, false, false);
      }
    }

    // epilogue: max over this tile's 16 t positions (mask t >= Sp)
    float mv = -3.4e38f;
#pragma unroll
    for (int i = 0; i < 8; ++i) {
      int t = tBase + i + ((lane < 16) ? 0 : 8);  // C layout: M = i + 8*half
      mv = (t < Sp) ? fmaxf(mv, acc[i]) : mv;
    }
    mv = fmaxf(mv, __shfl_xor(mv, 16, 32));       // wave32 cross-half max

    int f = fTile * 16 + m;
    if (lane < 16 && f < NFn) {
      float v = fmaxf(mv + bias[f], 0.0f);        // ReLU(max_t y + bias)
      // exact float max for non-negative values via uint bit-pattern max
      atomicMax((unsigned int*)(feats + (size_t)b * (3 * NFn) + foff + f),
                __float_as_uint(v));
    }
  }
}

// ---------------------------------------------------------------
// 5) FC [512,300]x[300,5] + bias + log_softmax (tiny, VALU)
// ---------------------------------------------------------------
__global__ void fc_logsoftmax_kernel(const float* __restrict__ feats,
                                     const float* __restrict__ Wfc,
                                     const float* __restrict__ bfc,
                                     float* __restrict__ out) {
  int b = blockIdx.x * blockDim.x + threadIdx.x;
  if (b >= Bn) return;
  float acc[NCLS];
#pragma unroll
  for (int c = 0; c < NCLS; ++c) acc[c] = bfc[c];
  for (int j = 0; j < 3 * NFn; ++j) {
    float xv = feats[(size_t)b * (3 * NFn) + j];
#pragma unroll
    for (int c = 0; c < NCLS; ++c) acc[c] += xv * Wfc[c * (3 * NFn) + j];
  }
  float mx = acc[0];
#pragma unroll
  for (int c = 1; c < NCLS; ++c) mx = fmaxf(mx, acc[c]);
  float s = 0.0f;
#pragma unroll
  for (int c = 0; c < NCLS; ++c) s += __expf(acc[c] - mx);
  float lse = mx + __logf(s);
#pragma unroll
  for (int c = 0; c < NCLS; ++c) out[(size_t)b * NCLS + c] = acc[c] - lse;
}

// ---------------------------------------------------------------
extern "C" void kernel_launch(void* const* d_in, const int* in_sizes, int n_in,
                              void* d_out, int out_size, void* d_ws, size_t ws_size,
                              hipStream_t stream) {
  const float* x   = (const float*)d_in[0];
  const float* w3  = (const float*)d_in[1];
  const float* b3  = (const float*)d_in[2];
  const float* w4  = (const float*)d_in[3];
  const float* b4  = (const float*)d_in[4];
  const float* w5  = (const float*)d_in[5];
  const float* b5  = (const float*)d_in[6];
  const float* Wfc = (const float*)d_in[7];
  const float* bfc = (const float*)d_in[8];
  float* out = (float*)d_out;

  // workspace carve-up
  const size_t xpadElems  = (size_t)Bn * S4 * EP;          // 21,626,880 bf16
  const size_t wfragElems = (size_t)12 * ECH * NFT * 512;  //    430,080 bf16
  __bf16* xpad  = (__bf16*)d_ws;
  __bf16* wfrag = xpad + xpadElems;
  float*  feats = (float*)(wfrag + wfragElems);            // 512*300 f32

  // 1) convert + pad x
  {
    int total = (int)xpadElems;
    cvt_x_kernel<<<(total + 255) / 256, 256, 0, stream>>>(x, xpad, total);
  }
  // 2) pack weight fragments
  pack_w_kernel<<<12 * ECH * NFT, 32, 0, stream>>>(w3, w4, w5, wfrag);
  // 3) zero pooled features
  {
    int n = Bn * 3 * NFn;
    zero_feats_kernel<<<(n + 255) / 256, 256, 0, stream>>>(feats, n);
  }
  // 4) WMMA conv + fused ReLU/maxpool, one instantiation per kernel size
  conv_wmma_kernel<3, 0, 0><<<dim3(Bn, 2), 256, 0, stream>>>(
      xpad, wfrag, b3, feats, 0);
  conv_wmma_kernel<4, 1, 3><<<dim3(Bn, 2), 256, 0, stream>>>(
      xpad, wfrag, b4, feats, NFn);
  conv_wmma_kernel<5, 2, 7><<<dim3(Bn, 2), 256, 0, stream>>>(
      xpad, wfrag, b5, feats, 2 * NFn);
  // 5) FC + log_softmax
  fc_logsoftmax_kernel<<<(Bn + 255) / 256, 256, 0, stream>>>(feats, Wfc, bfc, out);
}
